// InterpretableMultiHeadAttention_16810501997024
// MI455X (gfx1250) — compile-verified, matrix operand order checked by reading, not run on hardware
//
#include <hip/hip_runtime.h>
#include <hip/hip_bf16.h>

// ---------------------------------------------------------------------------
// InterpretableMultiHeadAttention for MI455X (gfx1250, wave32, WMMA).
// Pipeline:
//   0. transpose+convert weights to f16 (Wq^T, Wk^T, Wv^T, Wo^T)
//   1. Qh = q@Wq (f16), Kh = k@Wk (f16)      [WMMA GEMM, f16 out]
//   2. Vt  = (v@Wv)^T (f16, [64 x T] / batch) [WMMA GEMM, transposed f16 out]
//   3. pass1: per-head causal softmax stats (row max m, row sum Z) via WMMA
//   4. pass2: recompute scores, attn_avg = mean_h softmax -> d_out (f32)
//   5. tmp = attn_avg @ V                     [WMMA GEMM, f32 out]
//   6. out = tmp @ Wo                         [WMMA GEMM, f32 out]
// ---------------------------------------------------------------------------

typedef __attribute__((ext_vector_type(16))) _Float16 v16h;
typedef __attribute__((ext_vector_type(8)))  float    v8f;

#define WMMA_F16(a, b, c) \
  __builtin_amdgcn_wmma_f32_16x16x32_f16(false, (a), false, (b), (short)0, (c), false, false)

static constexpr int Bq = 2;
static constexpr int Ts = 2048;
static constexpr int Dm = 1024;
static constexpr int Hh = 16;
static constexpr int Dh = 64;

// --- fragment loaders -------------------------------------------------------
// 16-bit A/B fragment layout (16x32, per lane 16 halfs):
//   lane<16: K = kb+{0..7} in elems 0..7, K = kb+{16..23} in elems 8..15, kb=0
//   lane>=16: kb=8. Row (A) / column-as-row-of-Bt (B) = lane&15.
static __device__ __forceinline__ v16h load_frag16(const _Float16* __restrict__ src,
                                                   int ld, int row_base, int k0) {
  int lane = threadIdx.x & 31;
  int r = lane & 15;
  int kb = (lane & 16) ? 8 : 0;
  const _Float16* p = src + (size_t)(row_base + r) * ld + (k0 + kb);
  v16h f;
#pragma unroll
  for (int i = 0; i < 8; ++i) { f[i] = p[i]; f[8 + i] = p[16 + i]; }
  return f;
}

static __device__ __forceinline__ v16h load_frag32(const float* __restrict__ src,
                                                   int ld, int row_base, int k0) {
  int lane = threadIdx.x & 31;
  int r = lane & 15;
  int kb = (lane & 16) ? 8 : 0;
  const float* p = src + (size_t)(row_base + r) * ld + (k0 + kb);
  v16h f;
#pragma unroll
  for (int i = 0; i < 8; ++i) {
    f[i]     = (_Float16)p[i];
    f[8 + i] = (_Float16)p[16 + i];
  }
  return f;
}

// reductions within the 16-lane half (xor masks 1..8 stay inside each half)
static __device__ __forceinline__ float rmax16(float v) {
#pragma unroll
  for (int m = 1; m <= 8; m <<= 1) v = fmaxf(v, __shfl_xor(v, m, 32));
  return v;
}
static __device__ __forceinline__ float rsum16(float v) {
#pragma unroll
  for (int m = 1; m <= 8; m <<= 1) v += __shfl_xor(v, m, 32);
  return v;
}

// --- weight transpose+convert: Wt[n*K+k] = (f16) W[k*N+n] -------------------
__global__ void transpose_cvt_f16(const float* __restrict__ W, _Float16* __restrict__ Wt,
                                  int K, int N) {
  int idx = blockIdx.x * blockDim.x + threadIdx.x;
  if (idx >= K * N) return;
  int n = idx / K;
  int k = idx - n * K;
  Wt[(size_t)n * K + k] = (_Float16)W[(size_t)k * N + n];
}

// --- generic WMMA GEMM: D[M,N] = A_f32[M,K] * Bt_f16[N,K]^T -----------------
// MODE 0: f16 row-major out; MODE 1: f32 row-major out; MODE 2: f16 transposed
// out (out[col*ldT + row]).  Each wave: one 16x64 strip (1 A frag, 4 WMMAs/step).
template <int MODE>
__global__ void gemm_wmma(const float* __restrict__ A, const _Float16* __restrict__ Bt,
                          void* __restrict__ out, int M, int N, int K, int ldT) {
  int lane = threadIdx.x & 31;
  int wave = threadIdx.x >> 5;
  int row  = (blockIdx.x * 8 + wave) * 16;
  int nb   = blockIdx.y * 64;
  if (row >= M) return;

  v8f acc[4] = {};
  for (int k0 = 0; k0 < K; k0 += 32) {
    v16h a = load_frag32(A, K, row, k0);
#pragma unroll
    for (int j = 0; j < 4; ++j) {
      v16h b = load_frag16(Bt, K, nb + j * 16, k0);
      acc[j] = WMMA_F16(a, b, acc[j]);
    }
  }

  int half = (lane & 16) ? 8 : 0;
  int n    = lane & 15;
#pragma unroll
  for (int j = 0; j < 4; ++j) {
    int col = nb + j * 16 + n;
#pragma unroll
    for (int r = 0; r < 8; ++r) {
      int m = row + half + r;
      float v = acc[j][r];
      if (MODE == 0)      ((_Float16*)out)[(size_t)m * N + col] = (_Float16)v;
      else if (MODE == 1) ((float*)out)[(size_t)m * N + col]    = v;
      else                ((_Float16*)out)[(size_t)col * ldT + m] = (_Float16)v;
    }
  }
}

// --- pass 1: per (b,h,qtile) causal online-softmax stats --------------------
__global__ void attn_stats(const _Float16* __restrict__ Qh, const _Float16* __restrict__ Kh,
                           float* __restrict__ mstat, float* __restrict__ zstat) {
  int lane = threadIdx.x & 31;
  int wave = threadIdx.x >> 5;
  const int qtiles = Ts / 16;
  int task = blockIdx.x * 8 + wave;                 // (b*16+h)*qtiles + qt
  if (task >= Bq * Hh * qtiles) return;
  int qt = task % qtiles;
  int bh = task / qtiles;
  int h  = bh & 15;
  int b  = bh >> 4;

  int qbase = qt * 16;
  int d0    = h * Dh;
  const _Float16* Qb = Qh + (size_t)b * Ts * Dm;
  const _Float16* Kb = Kh + (size_t)b * Ts * Dm;

  v16h aq0 = load_frag16(Qb, Dm, qbase, d0);
  v16h aq1 = load_frag16(Qb, Dm, qbase, d0 + 32);

  int half = (lane & 16) ? 8 : 0;
  int n    = lane & 15;
  float mrow[8], zrow[8];
#pragma unroll
  for (int r = 0; r < 8; ++r) { mrow[r] = -1e30f; zrow[r] = 0.f; }

  for (int kt = 0; kt <= qt; ++kt) {
    int kbase = kt * 16;
    v16h bk0 = load_frag16(Kb, Dm, kbase, d0);
    v16h bk1 = load_frag16(Kb, Dm, kbase, d0 + 32);
    v8f s = {};
    s = WMMA_F16(aq0, bk0, s);
    s = WMMA_F16(aq1, bk1, s);
#pragma unroll
    for (int r = 0; r < 8; ++r) {
      int qrow = qbase + half + r;
      float sv = s[r] * 0.125f;                     // 1/sqrt(64)
      if (kbase + n > qrow) sv = -1e30f;            // causal mask
      float tm = rmax16(sv);
      float mn = fmaxf(mrow[r], tm);
      float p  = __expf(sv - mn);
      float ts = rsum16(p);
      zrow[r]  = zrow[r] * __expf(mrow[r] - mn) + ts;
      mrow[r]  = mn;
    }
  }
  if ((lane & 15) == 0) {
    size_t base = ((size_t)(b * Hh + h)) * Ts + qbase + half;
#pragma unroll
    for (int r = 0; r < 8; ++r) { mstat[base + r] = mrow[r]; zstat[base + r] = zrow[r]; }
  }
}

// --- pass 2: attn_avg tile = mean_h softmax_h, written to d_out -------------
__global__ void attn_write(const _Float16* __restrict__ Qh, const _Float16* __restrict__ Kh,
                           const float* __restrict__ mstat, const float* __restrict__ zstat,
                           float* __restrict__ attn) {
  int lane = threadIdx.x & 31;
  int wave = threadIdx.x >> 5;
  const int tiles = Ts / 16;
  int task = blockIdx.x * 8 + wave;                 // ((b*tiles)+qt)*tiles + kt
  if (task >= Bq * tiles * tiles) return;
  int kt = task % tiles;
  int t2 = task / tiles;
  int qt = t2 % tiles;
  int b  = t2 / tiles;

  int qbase = qt * 16, kbase = kt * 16;
  int half = (lane & 16) ? 8 : 0;
  int n    = lane & 15;
  float* attnB = attn + (size_t)b * Ts * Ts;

  if (kt > qt) {                                    // masked region -> exact zeros
#pragma unroll
    for (int r = 0; r < 8; ++r)
      attnB[(size_t)(qbase + half + r) * Ts + kbase + n] = 0.f;
    return;
  }

  const _Float16* Qb = Qh + (size_t)b * Ts * Dm;
  const _Float16* Kb = Kh + (size_t)b * Ts * Dm;
  float acc[8] = {0.f, 0.f, 0.f, 0.f, 0.f, 0.f, 0.f, 0.f};

  for (int h = 0; h < Hh; ++h) {
    int d0 = h * Dh;
    v16h aq0 = load_frag16(Qb, Dm, qbase, d0);
    v16h aq1 = load_frag16(Qb, Dm, qbase, d0 + 32);
    v16h bk0 = load_frag16(Kb, Dm, kbase, d0);
    v16h bk1 = load_frag16(Kb, Dm, kbase, d0 + 32);
    v8f s = {};
    s = WMMA_F16(aq0, bk0, s);
    s = WMMA_F16(aq1, bk1, s);
    size_t sb = ((size_t)(b * Hh + h)) * Ts + qbase + half;
#pragma unroll
    for (int r = 0; r < 8; ++r) {
      int qrow = qbase + half + r;
      if (kbase + n <= qrow) {
        float sv = s[r] * 0.125f;
        acc[r] += __expf(sv - mstat[sb + r]) / zstat[sb + r];
      }
    }
  }
#pragma unroll
  for (int r = 0; r < 8; ++r)
    attnB[(size_t)(qbase + half + r) * Ts + kbase + n] = acc[r] * 0.0625f; // /16 heads
}

// ---------------------------------------------------------------------------
extern "C" void kernel_launch(void* const* d_in, const int* in_sizes, int n_in,
                              void* d_out, int out_size, void* d_ws, size_t ws_size,
                              hipStream_t stream) {
  const float* q  = (const float*)d_in[0];
  const float* k  = (const float*)d_in[1];
  const float* v  = (const float*)d_in[2];
  // d_in[3] = mask (causal tril) — implemented analytically
  const float* Wq = (const float*)d_in[4];
  const float* Wk = (const float*)d_in[5];
  const float* Wv = (const float*)d_in[6];
  const float* Wo = (const float*)d_in[7];

  float* out_p  = (float*)d_out;                      // [B,T,Dm]
  float* attn_p = out_p + (size_t)Bq * Ts * Dm;       // [B,T,T]

  // ---- workspace carve (f16 regions first, then f32) ----
  char* w = (char*)d_ws;
  auto take = [&](size_t bytes) { char* p = w; w += (bytes + 255) & ~(size_t)255; return p; };
  _Float16* Wqt = (_Float16*)take((size_t)Dm * Dm * 2);
  _Float16* Wkt = (_Float16*)take((size_t)Dm * Dm * 2);
  _Float16* Wvt = (_Float16*)take((size_t)Dh * Dm * 2);
  _Float16* Wot = (_Float16*)take((size_t)Dm * Dh * 2);
  _Float16* Qh  = (_Float16*)take((size_t)Bq * Ts * Dm * 2);
  _Float16* Kh  = (_Float16*)take((size_t)Bq * Ts * Dm * 2);
  _Float16* Vt  = (_Float16*)take((size_t)Bq * Dh * Ts * 2);
  float* mstat  = (float*)take((size_t)Bq * Hh * Ts * 4);
  float* zstat  = (float*)take((size_t)Bq * Hh * Ts * 4);
  float* tmp    = (float*)take((size_t)Bq * Ts * Dh * 4);

  // ---- 0) weight transpose+convert ----
  {
    int n;
    n = Dm * Dm; transpose_cvt_f16<<<(n + 255) / 256, 256, 0, stream>>>(Wq, Wqt, Dm, Dm);
    n = Dm * Dm; transpose_cvt_f16<<<(n + 255) / 256, 256, 0, stream>>>(Wk, Wkt, Dm, Dm);
    n = Dm * Dh; transpose_cvt_f16<<<(n + 255) / 256, 256, 0, stream>>>(Wv, Wvt, Dm, Dh);
    n = Dh * Dm; transpose_cvt_f16<<<(n + 255) / 256, 256, 0, stream>>>(Wo, Wot, Dh, Dm);
  }

  // ---- 1) Q/K projections -> f16 [B*T, Dm] ----
  {
    dim3 grid(Bq * Ts / 128, Dm / 64);
    gemm_wmma<0><<<grid, 256, 0, stream>>>(q, Wqt, Qh, Bq * Ts, Dm, Dm, 0);
    gemm_wmma<0><<<grid, 256, 0, stream>>>(k, Wkt, Kh, Bq * Ts, Dm, Dm, 0);
  }

  // ---- 2) V projection, stored transposed f16 [Dh, T] per batch ----
  for (int b = 0; b < Bq; ++b) {
    dim3 grid(Ts / 128, 1);
    gemm_wmma<2><<<grid, 256, 0, stream>>>(v + (size_t)b * Ts * Dm, Wvt,
                                           Vt + (size_t)b * Dh * Ts, Ts, Dh, Dm, Ts);
  }

  // ---- 3) softmax stats (pass 1) ----
  {
    int tasks = Bq * Hh * (Ts / 16);
    attn_stats<<<(tasks + 7) / 8, 256, 0, stream>>>(Qh, Kh, mstat, zstat);
  }

  // ---- 4) attn_avg (pass 2) -> d_out ----
  {
    int tasks = Bq * (Ts / 16) * (Ts / 16);
    attn_write<<<(tasks + 7) / 8, 256, 0, stream>>>(Qh, Kh, mstat, zstat, attn_p);
  }

  // ---- 5) tmp = attn_avg @ V  (per batch; upper triangle is exact zeros) ----
  for (int b = 0; b < Bq; ++b) {
    dim3 grid(Ts / 128, 1);
    gemm_wmma<1><<<grid, 256, 0, stream>>>(attn_p + (size_t)b * Ts * Ts,
                                           Vt + (size_t)b * Dh * Ts,
                                           tmp + (size_t)b * Ts * Dh, Ts, Dh, Ts, 0);
  }

  // ---- 6) out = tmp @ Wo ----
  {
    dim3 grid(Bq * Ts / 128, Dm / 64);
    gemm_wmma<1><<<grid, 256, 0, stream>>>(tmp, Wot, out_p, Bq * Ts, Dm, Dh, 0);
  }
}
